// fast_BEV_10488310137172
// MI455X (gfx1250) — matrix operand order repaired and checked by reading, not compiled, not run on hardware
//
#include <hip/hip_runtime.h>
#include <hip/hip_bf16.h>

typedef float v8f __attribute__((ext_vector_type(8)));
typedef __bf16 bf16x16 __attribute__((ext_vector_type(16)));
typedef unsigned short u16;
typedef u16 u16x8 __attribute__((ext_vector_type(8)));
typedef u16 u16x16 __attribute__((ext_vector_type(16)));

// async global->LDS builtin takes typed vector pointers (per hipcc diagnostic):
// param 0: AS(1) int __vector_size(8)*   param 1: AS(3) int __vector_size(8)*
typedef int v2i_vs __attribute__((vector_size(8)));
typedef __attribute__((address_space(1))) v2i_vs* gptr64;
typedef __attribute__((address_space(3))) v2i_vs* lptr64;

#define BATCH 2
#define NCAMS 6
#define CH 256
#define IH 32
#define IW 88
#define NXg 200
#define NYg 200
#define NZg 4
#define NPOS (NXg * NYg)      /* 40000 spatial positions per batch */
#define NVOX (NPOS * NZg)     /* 160000 voxels per batch */
#define OC 80
#define KTOT 1024             /* NZg * CH */
#define STRIDE_PX 8.0f        /* ceil(704/88) */
#define MT 128                /* spatial positions per block */
#define KC 64                 /* K chunk staged in LDS */
#define LDH 72                /* LDS row stride in halves (144B, 16B aligned) */

#define HAS_ASYNC_LDS __has_builtin(__builtin_amdgcn_global_load_async_to_lds_b64)

static __device__ __forceinline__ u16 f2bf(float f) {
  unsigned int u = __float_as_uint(f);
  u += 0x7FFFu + ((u >> 16) & 1u);   // round-to-nearest-even
  return (u16)(u >> 16);
}

static __device__ __forceinline__ void wait_async0() {
#if HAS_ASYNC_LDS
#if __has_builtin(__builtin_amdgcn_s_wait_asynccnt)
  __builtin_amdgcn_s_wait_asynccnt(0);
#else
  asm volatile("s_wait_asynccnt 0x0" ::: "memory");
#endif
#endif
}

// load 16 contiguous halves (unused; the B fragment is loaded with ldfrag_split)
static __device__ __forceinline__ bf16x16 ldfrag_contig(const u16* p) {
  u16x8 lo = *(const u16x8*)p;
  u16x8 hi = *(const u16x8*)(p + 8);
  u16x16 c = __builtin_shufflevector(lo, hi,
      0,1,2,3,4,5,6,7,8,9,10,11,12,13,14,15);
  return __builtin_bit_cast(bf16x16, c);
}

// load two 16B runs (A fragment: K = (e/8)*16 + (l/16)*8 + e%8)
static __device__ __forceinline__ bf16x16 ldfrag_split(const u16* p, int halfsel) {
  u16x8 lo = *(const u16x8*)(p + halfsel * 8);
  u16x8 hi = *(const u16x8*)(p + 16 + halfsel * 8);
  u16x16 c = __builtin_shufflevector(lo, hi,
      0,1,2,3,4,5,6,7,8,9,10,11,12,13,14,15);
  return __builtin_bit_cast(bf16x16, c);
}

// reuse_b hint: same B operand feeds the next WMMA in the burst
template <bool RB>
static __device__ __forceinline__ v8f do_wmma(bf16x16 a, bf16x16 b, v8f c) {
  return __builtin_amdgcn_wmma_f32_16x16x32_bf16(
      false, a, false, b, (short)0, c, false, RB);
}

// ---------------------------------------------------------------------------
// 1) proj[b,cam] = (img_aug @ lidar2image @ lidar_aug)[:3,:]   (12 matrices)
// ---------------------------------------------------------------------------
__global__ void bev_proj_kernel(const float* __restrict__ l2i,
                                const float* __restrict__ iam,
                                const float* __restrict__ lam,
                                float* __restrict__ proj) {
  int i = threadIdx.x;
  if (i >= BATCH * NCAMS) return;
  int b = i / NCAMS;
  const float* A = iam + i * 16;
  const float* M = l2i + i * 16;
  const float* L = lam + b * 16;
  float T[16];
  #pragma unroll
  for (int r = 0; r < 4; ++r)
    #pragma unroll
    for (int c = 0; c < 4; ++c) {
      float s = 0.f;
      #pragma unroll
      for (int k = 0; k < 4; ++k) s += A[r * 4 + k] * M[k * 4 + c];
      T[r * 4 + c] = s;
    }
  #pragma unroll
  for (int r = 0; r < 3; ++r)
    #pragma unroll
    for (int c = 0; c < 4; ++c) {
      float s = 0.f;
      #pragma unroll
      for (int k = 0; k < 4; ++k) s += T[r * 4 + k] * L[k * 4 + c];
      proj[i * 12 + r * 4 + c] = s;
    }
}

// ---------------------------------------------------------------------------
// 2) Fold BN into weights (bf16) and bias (f32).
// ---------------------------------------------------------------------------
__global__ void bev_weights_kernel(const float* __restrict__ w,
                                   const float* __restrict__ cb,
                                   const float* __restrict__ gam,
                                   const float* __restrict__ bet,
                                   const float* __restrict__ mn,
                                   const float* __restrict__ vr,
                                   u16* __restrict__ wbf,
                                   float* __restrict__ bias) {
  int i = blockIdx.x * 256 + threadIdx.x;
  if (i < OC * KTOT) {
    int o = i >> 10;
    float inv = gam[o] / sqrtf(vr[o] + 1e-5f);
    wbf[i] = f2bf(w[i] * inv);
  }
  if (i < OC) {
    float inv = gam[i] / sqrtf(vr[i] + 1e-5f);
    bias[i] = cb[i] * inv + bet[i] - mn[i] * inv;
  }
}

// ---------------------------------------------------------------------------
// 3) Per-voxel winner camera -> gather base offset (or -1)
// ---------------------------------------------------------------------------
__global__ void bev_index_kernel(const float* __restrict__ proj,
                                 const float* __restrict__ vg,
                                 int* __restrict__ vidx) {
  int i = blockIdx.x * blockDim.x + threadIdx.x;
  if (i >= BATCH * NVOX) return;
  int b = i / NVOX;
  int n = i - b * NVOX;
  float gx = vg[n];
  float gy = vg[NVOX + n];
  float gz = vg[2 * NVOX + n];
  int bestoff = -1;
  #pragma unroll
  for (int cam = 0; cam < NCAMS; ++cam) {
    const float* P = proj + (b * NCAMS + cam) * 12;
    float X = P[0] * gx + P[1] * gy + P[2]  * gz + P[3];
    float Y = P[4] * gx + P[5] * gy + P[6]  * gz + P[7];
    float Z = P[8] * gx + P[9] * gy + P[10] * gz + P[11];
    float iz = 1.0f / Z;
    int u = (int)rintf(X * iz * (1.0f / STRIDE_PX));
    int v = (int)rintf(Y * iz * (1.0f / STRIDE_PX));
    bool valid = (Z > 0.f) & (u >= 0) & (v >= 0) & (u < IW) & (v < IH);
    if (valid) bestoff = cam * (CH * IH * IW) + v * IW + u;   // highest cam wins
  }
  vidx[i] = bestoff;
}

// ---------------------------------------------------------------------------
// 4) Fused gather -> bf16 WMMA GEMM -> BN/ReLU epilogue.
// ---------------------------------------------------------------------------
__global__ __launch_bounds__(256) void bev_gemm_kernel(
    const float* __restrict__ feat,   // [B, NCAM, CH, IH, IW]
    const int* __restrict__ vidx,     // [B, NVOX]
    const u16* __restrict__ wbf,      // [OC, KTOT] bf16 (BN-folded)
    const float* __restrict__ bias,   // [OC]
    float* __restrict__ out)          // [B, OC, NPOS]
{
  __shared__ u16 lds_act[MT * LDH];   // 18432 B
  __shared__ u16 lds_w[OC * LDH];     // 11520 B

  const int t = threadIdx.x;
  const int b = blockIdx.y;
  const int p0 = blockIdx.x * MT;

  const float* featb = feat + (size_t)b * NCAMS * CH * IH * IW;
  const int* idxb = vidx + b * NVOX;

  const int lane = t & 31;
  const int wv = t >> 5;             // wave id 0..7 -> spatial sub-tile
  const int halfsel = lane >> 4;
  const int l16 = lane & 15;

  v8f acc[5];
  #pragma unroll
  for (int i = 0; i < 5; ++i) acc[i] = (v8f){0.f,0.f,0.f,0.f,0.f,0.f,0.f,0.f};

  const int p_loc = t & (MT - 1);    // lanes run along adjacent voxels
  const int cpart = t >> 7;          // 0..1 -> which 32-channel half
  const int p = p0 + p_loc;
  const int HW = IH * IW;

  for (int k0 = 0; k0 < KTOT; k0 += KC) {
    const int z = k0 >> 8;                 // k = z*256 + c
    const int cbase = (k0 & 255) + cpart * 32;

    // ---- stage activations: batched gather, cvt bf16, packed LDS stores ----
    int off = (p < NPOS) ? idxb[p * NZg + z] : -1;
    u16* dst = &lds_act[p_loc * LDH + cpart * 32];
    u16 h[32];
    if (off >= 0) {
      const float* src = featb + off + cbase * HW;
      float tmp[32];
      #pragma unroll
      for (int j = 0; j < 32; ++j) tmp[j] = src[j * HW];
      #pragma unroll
      for (int j = 0; j < 32; ++j) h[j] = f2bf(tmp[j]);
    } else {
      #pragma unroll
      for (int j = 0; j < 32; ++j) h[j] = 0;
    }
    #pragma unroll
    for (int q = 0; q < 4; ++q) {
      u16x8 v;
      #pragma unroll
      for (int j = 0; j < 8; ++j) v[j] = h[q * 8 + j];
      *(u16x8*)(dst + q * 8) = v;
    }

    // ---- stage weight chunk: 80 x 64 halves ----
#if HAS_ASYNC_LDS
    #pragma unroll
    for (int i5 = 0; i5 < 5; ++i5) {          // 1280 / 256 = 5 exact
      int i = i5 * 256 + t;
      int o = i >> 4;
      int kk4 = (i & 15) * 4;
      __builtin_amdgcn_global_load_async_to_lds_b64(
          (gptr64)(wbf + o * KTOT + k0 + kk4),
          (lptr64)(&lds_w[o * LDH + kk4]),
          0, 0);
    }
#else
    for (int i = t; i < (OC * KC) / 4; i += 256) {
      int o = i >> 4;
      int kk4 = (i & 15) * 4;
      *(uint2*)(&lds_w[o * LDH + kk4]) =
          *(const uint2*)(wbf + o * KTOT + k0 + kk4);
    }
#endif
    // warm next weight chunk (global_prefetch_b8)
    if ((k0 + KC < KTOT) && t < OC)
      __builtin_prefetch(&wbf[t * KTOT + k0 + KC], 0, 3);

    wait_async0();
    __syncthreads();

    // ---- compute: hoist all fragments, then WMMA bursts with reuse_b ----
    #pragma unroll
    for (int ks = 0; ks < 2; ++ks) {
      const u16* bp = &lds_act[(wv * 16 + l16) * LDH + ks * 32];
      bf16x16 bfrag = ldfrag_split(bp, halfsel);
      bf16x16 af[5];
      #pragma unroll
      for (int osub = 0; osub < 5; ++osub)
        af[osub] = ldfrag_split(&lds_w[(osub * 16 + l16) * LDH + ks * 32],
                                halfsel);
      acc[0] = do_wmma<true >(af[0], bfrag, acc[0]);
      acc[1] = do_wmma<true >(af[1], bfrag, acc[1]);
      acc[2] = do_wmma<true >(af[2], bfrag, acc[2]);
      acc[3] = do_wmma<true >(af[3], bfrag, acc[3]);
      acc[4] = do_wmma<false>(af[4], bfrag, acc[4]);
    }
    __syncthreads();
  }

  // ---- epilogue: D layout f32 16x16: VGPR r -> M=o=r+(l/16)*8, N=p=l%16 ----
  const int pp = p0 + wv * 16 + l16;
  if (pp < NPOS) {
    #pragma unroll
    for (int osub = 0; osub < 5; ++osub) {
      #pragma unroll
      for (int r = 0; r < 8; ++r) {
        int o = osub * 16 + halfsel * 8 + r;
        float v = acc[osub][r] + bias[o];
        out[((size_t)(b * OC + o)) * NPOS + pp] = v > 0.f ? v : 0.f;
      }
    }
  }
}

// ---------------------------------------------------------------------------
extern "C" void kernel_launch(void* const* d_in, const int* in_sizes, int n_in,
                              void* d_out, int out_size, void* d_ws, size_t ws_size,
                              hipStream_t stream) {
  (void)in_sizes; (void)n_in; (void)out_size; (void)ws_size;
  const float* feat = (const float*)d_in[0];
  const float* l2i  = (const float*)d_in[1];
  const float* iam  = (const float*)d_in[2];
  const float* lam  = (const float*)d_in[3];
  const float* vg   = (const float*)d_in[4];
  const float* cw   = (const float*)d_in[5];
  const float* cb   = (const float*)d_in[6];
  const float* bg   = (const float*)d_in[7];
  const float* bb   = (const float*)d_in[8];
  const float* bm   = (const float*)d_in[9];
  const float* bv   = (const float*)d_in[10];
  float* out = (float*)d_out;

  char* ws = (char*)d_ws;
  float* proj = (float*)ws;                          // 144 floats
  float* bias = (float*)(ws + 1024);                 // 80 floats
  u16*   wbf  = (u16*)(ws + 4096);                   // 80*1024 bf16 = 160KB
  int*   vidx = (int*)(ws + 4096 + OC * KTOT * 2);   // 320000 int32 = 1.28MB

  bev_proj_kernel<<<1, 16, 0, stream>>>(l2i, iam, lam, proj);
  bev_weights_kernel<<<(OC * KTOT + 255) / 256, 256, 0, stream>>>(
      cw, cb, bg, bb, bm, bv, wbf, bias);
  bev_index_kernel<<<(BATCH * NVOX + 255) / 256, 256, 0, stream>>>(
      proj, vg, vidx);

  dim3 grid((NPOS + MT - 1) / MT, BATCH);
  bev_gemm_kernel<<<grid, 256, 0, stream>>>(feat, vidx, wbf, bias, out);
}
